// VisionMamba_61744449847376
// MI455X (gfx1250) — matrix operand order repaired and checked by reading, not compiled, not run on hardware
//
#include <hip/hip_runtime.h>

// ---------------- problem constants ----------------
constexpr int B0    = 4;
constexpr int LTOK  = 2048;           // 16*16*8 tokens
constexpr int M     = B0 * LTOK;      // 8192 rows
constexpr int E     = 384;
constexpr int E2    = 768;
constexpr int NST   = 16;
constexpr int DTR   = 24;
constexpr int DBLW  = 64;             // 56 padded -> 64
constexpr int DEPTH = 8;
constexpr int KPATCH= 64;             // 4*4*4 patch
constexpr int KDT   = 32;             // 24 padded -> 32

typedef __attribute__((ext_vector_type(16))) __bf16 v16bf;
typedef __attribute__((ext_vector_type(8)))  __bf16 v8bf;
typedef __attribute__((ext_vector_type(8)))  float  v8f;

union AFrag { v16bf v; v8bf h[2]; };

// ---------------- bf16 convert (with optional row/col zero-padding) --------
__global__ __launch_bounds__(256) void cvt_pad_kernel(
    const float* __restrict__ in, __bf16* __restrict__ out,
    int nsrc, int ksrc, int kp, int total)
{
    int i = blockIdx.x * blockDim.x + threadIdx.x;
    if (i >= total) return;
    int r = i / kp, c = i - r * kp;
    float v = (r < nsrc && c < ksrc) ? in[(size_t)r * ksrc + c] : 0.f;
    out[i] = (__bf16)v;
}

// ---------------- patch gather (im2col) to bf16 ----------------------------
// x: (4,1,64,64,32), out: (M, 64)   l = (gx*16+gy)*8+gz, kk = i*16+j*4+k
__global__ __launch_bounds__(256) void im2col_kernel(
    const float* __restrict__ x, __bf16* __restrict__ out)
{
    int idx = blockIdx.x * blockDim.x + threadIdx.x;
    if (idx >= M * KPATCH) return;
    int m = idx >> 6, kk = idx & 63;
    int b = m >> 11, l = m & 2047;
    int gx = l >> 7, gy = (l >> 3) & 15, gz = l & 7;
    int i = kk >> 4, j = (kk >> 2) & 3, kc = kk & 3;
    size_t src = ((size_t)b * 64 + (gx * 4 + i)) * 2048
               + (size_t)(gy * 4 + j) * 32 + (gz * 4 + kc);
    out[idx] = (__bf16)x[src];
}

// ---------------- WMMA GEMM: D(MxN,f32) = A(MxK,bf16) * W(NxK,bf16)^T ------
// wave32 computes a 32x64 tile: 2 A-fragments x 4 B-fragments ->
// 8x v_wmma_f32_16x16x32_bf16 per 32-deep K-step.
// EPI: 0 plain | 1 +bias+pos(extra) | 2 residual add into D | 3 softplus(acc+bias)
//      4 plain + dual store (f32 D and bf16 Dbf)
template <int EPI>
__global__ __launch_bounds__(256) void gemm_wmma_kernel(
    const __bf16* __restrict__ A, const __bf16* __restrict__ W,
    float* __restrict__ D, __bf16* __restrict__ Dbf,
    const float* __restrict__ bias, const float* __restrict__ extra,
    int Mr, int N, int K)
{
    const int wid  = (blockIdx.x * blockDim.x + threadIdx.x) >> 5;
    const int lane = threadIdx.x & 31;
    const int nblk = N >> 6;
    const int mt = wid / nblk;            // 32-row tile index
    const int nt = wid - mt * nblk;       // 64-col tile index
    if (mt >= (Mr >> 5)) return;

    const int half  = lane >> 4;          // which K-half this lane owns
    const int lq    = lane & 15;
    const int nbase = nt << 6;

    const __bf16* arow0 = A + (size_t)((mt << 5) + lq)      * K;
    const __bf16* arow1 = A + (size_t)((mt << 5) + 16 + lq) * K;
    const __bf16* w0    = W + (size_t)(nbase + lq) * K;
    const size_t  wstr  = (size_t)16 * K;

    v8f acc[2][4] = {{v8f{}, v8f{}, v8f{}, v8f{}},
                     {v8f{}, v8f{}, v8f{}, v8f{}}};
    for (int k0 = 0; k0 < K; k0 += 32) {
        if (k0 + 64 < K) {  // pull the K-block two steps ahead into WGP$
            __builtin_prefetch(arow0 + k0 + 64, 0, 3);
            __builtin_prefetch(arow1 + k0 + 64, 0, 3);
            __builtin_prefetch(w0 + 0 * wstr + k0 + 64, 0, 3);
            __builtin_prefetch(w0 + 1 * wstr + k0 + 64, 0, 3);
            __builtin_prefetch(w0 + 2 * wstr + k0 + 64, 0, 3);
            __builtin_prefetch(w0 + 3 * wstr + k0 + 64, 0, 3);
        }
        AFrag a0, a1;
        a0.v = *(const v16bf*)(arow0 + k0 + half * 16);
        a1.v = *(const v16bf*)(arow1 + k0 + half * 16);
        const int kb = k0 + half * 16;
        v16bf b0 = *(const v16bf*)(w0 + 0 * wstr + kb);
        v16bf b1 = *(const v16bf*)(w0 + 1 * wstr + kb);
        v16bf b2 = *(const v16bf*)(w0 + 2 * wstr + kb);
        v16bf b3 = *(const v16bf*)(w0 + 3 * wstr + kb);
        acc[0][0] = __builtin_amdgcn_wmma_f32_16x16x32_bf16(false, a0.v, false, b0, (short)0, acc[0][0], false, false);
        acc[0][1] = __builtin_amdgcn_wmma_f32_16x16x32_bf16(false, a0.v, false, b1, (short)0, acc[0][1], false, false);
        acc[0][2] = __builtin_amdgcn_wmma_f32_16x16x32_bf16(false, a0.v, false, b2, (short)0, acc[0][2], false, false);
        acc[0][3] = __builtin_amdgcn_wmma_f32_16x16x32_bf16(false, a0.v, false, b3, (short)0, acc[0][3], false, false);
        acc[1][0] = __builtin_amdgcn_wmma_f32_16x16x32_bf16(false, a1.v, false, b0, (short)0, acc[1][0], false, false);
        acc[1][1] = __builtin_amdgcn_wmma_f32_16x16x32_bf16(false, a1.v, false, b1, (short)0, acc[1][1], false, false);
        acc[1][2] = __builtin_amdgcn_wmma_f32_16x16x32_bf16(false, a1.v, false, b2, (short)0, acc[1][2], false, false);
        acc[1][3] = __builtin_amdgcn_wmma_f32_16x16x32_bf16(false, a1.v, false, b3, (short)0, acc[1][3], false, false);
    }

#pragma unroll
    for (int i = 0; i < 2; ++i) {
#pragma unroll
        for (int j = 0; j < 4; ++j) {
            const int ncol = nbase + j * 16 + lq;
            float bsv = 0.f;
            if constexpr (EPI == 1 || EPI == 3) bsv = bias[ncol];
#pragma unroll
            for (int r = 0; r < 8; ++r) {
                const int row = (mt << 5) + i * 16 + half * 8 + r;
                const size_t oi = (size_t)row * N + ncol;
                float v = acc[i][j][r];
                if constexpr (EPI == 1) v += bsv + extra[(size_t)(row & (LTOK - 1)) * N + ncol];
                if constexpr (EPI == 2) v += D[oi];
                if constexpr (EPI == 3) { v += bsv; v = (v > 20.f) ? v : log1pf(__expf(v)); }
                D[oi] = v;
                if constexpr (EPI == 4) Dbf[oi] = (__bf16)v;
            }
        }
    }
}

// ---------------- LayerNorm: one wave32 per row of 384 ---------------------
template <typename OutT>
__global__ __launch_bounds__(256) void ln_kernel(
    const float* __restrict__ in, const float* __restrict__ g,
    const float* __restrict__ be, OutT* __restrict__ out, int rows)
{
    int wid  = (blockIdx.x * blockDim.x + threadIdx.x) >> 5;
    int lane = threadIdx.x & 31;
    if (wid >= rows) return;
    const float* r = in + (size_t)wid * E;
    float s = 0.f, s2 = 0.f;
    for (int i = lane; i < E; i += 32) { float v = r[i]; s += v; s2 += v * v; }
#pragma unroll
    for (int off = 16; off > 0; off >>= 1) {
        s  += __shfl_xor(s,  off, 32);
        s2 += __shfl_xor(s2, off, 32);
    }
    float mean = s * (1.f / E);
    float var  = s2 * (1.f / E) - mean * mean;
    float inv  = rsqrtf(var + 1e-5f);
    OutT* o = out + (size_t)wid * E;
    for (int i = lane; i < E; i += 32)
        o[i] = (OutT)((r[i] - mean) * inv * g[i] + be[i]);
}

// ---------------- causal depthwise conv (k=4) + SiLU -----------------------
__global__ __launch_bounds__(256) void conv_silu_kernel(
    const float* __restrict__ xz, const float* __restrict__ cw,
    const float* __restrict__ cb, float* __restrict__ xc,
    __bf16* __restrict__ xcb)
{
    int idx = blockIdx.x * blockDim.x + threadIdx.x;
    if (idx >= M * E) return;
    int m = idx / E, e = idx - m * E;
    int b = m / LTOK, l = m - b * LTOK;
    float acc = cb[e];
#pragma unroll
    for (int t = 0; t < 4; ++t) {
        int ls = l - 3 + t;
        if (ls >= 0) acc += xz[(size_t)(b * LTOK + ls) * E2 + e] * cw[e * 4 + t];
    }
    float s = acc / (1.f + __expf(-acc));
    xc[idx]  = s;
    xcb[idx] = (__bf16)s;
}

// ---------------- selective scan: 16 lanes per (b,e) channel ---------------
__global__ __launch_bounds__(256) void scan_kernel(
    const float* __restrict__ dt, const float* __restrict__ dbl,
    const float* __restrict__ xc, const float* __restrict__ xz,
    const float* __restrict__ a_log, const float* __restrict__ d_skip,
    __bf16* __restrict__ ybuf)
{
    int gid = blockIdx.x * blockDim.x + threadIdx.x;
    int ch = gid >> 4;          // (b,e) channel
    int n  = gid & 15;          // state index
    if (ch >= B0 * E) return;
    int b = ch / E, e = ch - b * E;
    float Ac = -__expf(a_log[e * NST + n]);
    float Dv = d_skip[e];
    float h = 0.f;
    const int base = b * LTOK;
    for (int l = 0; l < LTOK; ++l) {
        int m = base + l;
        if (l + 8 < LTOK) {  // stream next steps into WGP$ ahead of recurrence
            __builtin_prefetch(dt  + (size_t)(m + 8) * E + e, 0, 3);
            __builtin_prefetch(dbl + (size_t)(m + 8) * DBLW + DTR + n, 0, 3);
            __builtin_prefetch(xc  + (size_t)(m + 8) * E + e, 0, 3);
        }
        float dtv = dt[(size_t)m * E + e];
        float xv  = xc[(size_t)m * E + e];
        float Bv  = dbl[(size_t)m * DBLW + DTR + n];
        float Cv  = dbl[(size_t)m * DBLW + DTR + NST + n];
        h = h * __expf(dtv * Ac) + dtv * xv * Bv;
        float y = h * Cv;
        y += __shfl_xor(y, 8, 16);
        y += __shfl_xor(y, 4, 16);
        y += __shfl_xor(y, 2, 16);
        y += __shfl_xor(y, 1, 16);
        if (n == 0) {
            float zv  = xz[(size_t)m * E2 + E + e];
            float out = (y + xv * Dv) * (zv / (1.f + __expf(-zv)));
            ybuf[(size_t)m * E + e] = (__bf16)out;
        }
    }
}

// ---------------- host orchestration ---------------------------------------
static inline int nblk(long long threads) { return (int)((threads + 255) / 256); }
static inline int gemm_blocks(int m, int n) {
    long long waves = (long long)(m / 32) * (n / 64);
    return (int)((waves * 32 + 255) / 256);
}

extern "C" void kernel_launch(void* const* d_in, const int* in_sizes, int n_in,
                              void* d_out, int out_size, void* d_ws, size_t ws_size,
                              hipStream_t stream) {
    const float* x       = (const float*)d_in[0];
    const float* pos     = (const float*)d_in[1];
    const float* patch_w = (const float*)d_in[2];
    const float* patch_b = (const float*)d_in[3];
    const float* ln_w    = (const float*)d_in[4];
    const float* ln_b    = (const float*)d_in[5];
    const float* in_w    = (const float*)d_in[6];
    const float* conv_w  = (const float*)d_in[7];
    const float* conv_b  = (const float*)d_in[8];
    const float* xp_w    = (const float*)d_in[9];
    const float* dt_w    = (const float*)d_in[10];
    const float* dt_bias = (const float*)d_in[11];
    const float* A_log   = (const float*)d_in[12];
    const float* D_skip  = (const float*)d_in[13];
    const float* out_w   = (const float*)d_in[14];
    const float* fn_w    = (const float*)d_in[15];
    const float* fn_b    = (const float*)d_in[16];
    float* out = (float*)d_out;

    char* p = (char*)d_ws;
    auto carve = [&](size_t bytes) -> void* {
        void* r = (void*)p;
        p += (bytes + 255) & ~(size_t)255;
        return r;
    };
    float*  tok    = (float*) carve((size_t)M * E  * 4);
    __bf16* xln    = (__bf16*)carve((size_t)M * E  * 2);
    float*  xz     = (float*) carve((size_t)M * E2 * 4);
    float*  xc     = (float*) carve((size_t)M * E  * 4);
    __bf16* xcb    = (__bf16*)carve((size_t)M * E  * 2);
    float*  dbl    = (float*) carve((size_t)M * DBLW * 4);
    __bf16* dblb   = (__bf16*)carve((size_t)M * DBLW * 2);
    float*  dtv    = (float*) carve((size_t)M * E  * 4);
    __bf16* ybuf   = (__bf16*)carve((size_t)M * E  * 2);
    __bf16* patchA = (__bf16*)carve((size_t)M * KPATCH * 2);
    __bf16* wPatch = (__bf16*)carve((size_t)E * KPATCH * 2);
    __bf16* wIn    = (__bf16*)carve((size_t)DEPTH * E2 * E * 2);
    __bf16* wOut   = (__bf16*)carve((size_t)DEPTH * E  * E * 2);
    __bf16* wXp    = (__bf16*)carve((size_t)DEPTH * DBLW * E * 2);
    __bf16* wDt    = (__bf16*)carve((size_t)DEPTH * E * KDT * 2);

    // ---- weight conversion (bf16, (N,K) row-major; pad x_proj rows / dt cols)
    cvt_pad_kernel<<<nblk(E * KPATCH), 256, 0, stream>>>(patch_w, wPatch, E, KPATCH, KPATCH, E * KPATCH);
    for (int d = 0; d < DEPTH; ++d) {
        cvt_pad_kernel<<<nblk(E2 * E), 256, 0, stream>>>(in_w  + (size_t)d * E2 * E,  wIn  + (size_t)d * E2 * E,  E2, E, E, E2 * E);
        cvt_pad_kernel<<<nblk(E * E),  256, 0, stream>>>(out_w + (size_t)d * E * E,   wOut + (size_t)d * E * E,   E,  E, E, E * E);
        cvt_pad_kernel<<<nblk(DBLW * E), 256, 0, stream>>>(xp_w + (size_t)d * 56 * E, wXp + (size_t)d * DBLW * E, 56, E, E, DBLW * E);
        cvt_pad_kernel<<<nblk(E * KDT), 256, 0, stream>>>(dt_w + (size_t)d * E * DTR, wDt + (size_t)d * E * KDT,  E, DTR, KDT, E * KDT);
    }

    // ---- patch embedding: tok = im2col(x) @ patch_w^T + patch_b + pos
    im2col_kernel<<<nblk((long long)M * KPATCH), 256, 0, stream>>>(x, patchA);
    gemm_wmma_kernel<1><<<gemm_blocks(M, E), 256, 0, stream>>>(
        patchA, wPatch, tok, nullptr, patch_b, pos, M, E, KPATCH);

    // ---- 8 Mamba layers
    for (int d = 0; d < DEPTH; ++d) {
        ln_kernel<__bf16><<<nblk((long long)M * 32), 256, 0, stream>>>(
            tok, ln_w + d * E, ln_b + d * E, xln, M);
        gemm_wmma_kernel<0><<<gemm_blocks(M, E2), 256, 0, stream>>>(
            xln, wIn + (size_t)d * E2 * E, xz, nullptr, nullptr, nullptr, M, E2, E);
        conv_silu_kernel<<<nblk((long long)M * E), 256, 0, stream>>>(
            xz, conv_w + (size_t)d * E * 4, conv_b + d * E, xc, xcb);
        gemm_wmma_kernel<4><<<gemm_blocks(M, DBLW), 256, 0, stream>>>(
            xcb, wXp + (size_t)d * DBLW * E, dbl, dblb, nullptr, nullptr, M, DBLW, E);
        gemm_wmma_kernel<3><<<gemm_blocks(M, E), 256, 0, stream>>>(
            dblb, wDt + (size_t)d * E * KDT, dtv, nullptr, dt_bias + d * E, nullptr, M, E, KDT);
        scan_kernel<<<nblk((long long)B0 * E * 16), 256, 0, stream>>>(
            dtv, dbl, xc, xz, A_log + (size_t)d * E * NST, D_skip + d * E, ybuf);
        gemm_wmma_kernel<2><<<gemm_blocks(M, E), 256, 0, stream>>>(
            ybuf, wOut + (size_t)d * E * E, tok, nullptr, nullptr, nullptr, M, E, E);
    }

    // ---- final LayerNorm -> f32 output
    ln_kernel<float><<<nblk((long long)M * 32), 256, 0, stream>>>(tok, fn_w, fn_b, out, M);

    (void)in_sizes; (void)n_in; (void)out_size; (void)ws_size;
}
